// Encoder_45689862095561
// MI455X (gfx1250) — matrix operand order repaired and checked by reading, not compiled, not run on hardware
//
#include <hip/hip_runtime.h>
#include <hip/hip_bf16.h>
#include <stdint.h>

// ---------------------------------------------------------------------------
// EPA encoder block for MI455X (gfx1250, wave32, WMMA).
// B=4, C=32, H=W=D=32, N=32768, P=64, heads=4, hd=8.
// ---------------------------------------------------------------------------

typedef __attribute__((ext_vector_type(16))) _Float16 v16h;
typedef __attribute__((ext_vector_type(8)))  float    v8f;

union V16H { v16h v; _Float16 h[16]; uint4 q[2]; };

__device__ __forceinline__ v8f wmma16(v16h a, v16h b, v8f c) {
  return __builtin_amdgcn_wmma_f32_16x16x32_f16(false, a, false, b, (short)0, c, false, false);
}

__device__ __forceinline__ float lrelu(float v) { return v > 0.f ? v : 0.01f * v; }

constexpr int NB   = 4;       // batch
constexpr int CC   = 32;      // channels
constexpr int NN   = 32768;   // tokens
constexpr int PP   = 64;      // proj size

// ===========================================================================
// K0: pack weights into WMMA B-matrix lane layout, fold BN into conv weights,
//     zero the split-K projection accumulator.
// B layout (16x16x32 f16): lane<16 -> n=lane, k=(e<8?e:8+e) ; lane>=16 -> n=lane-16, k=8+(e<8?e:8+e)
// ===========================================================================
__global__ __launch_bounds__(256) void enc_k0_pack(
    const float* __restrict__ Wq,
    const float* __restrict__ c1w, const float* __restrict__ c1b,
    const float* __restrict__ bn1g, const float* __restrict__ bn1b,
    const float* __restrict__ bn1m, const float* __restrict__ bn1v,
    const float* __restrict__ c2w, const float* __restrict__ c2b,
    const float* __restrict__ bn2g, const float* __restrict__ bn2b,
    const float* __restrict__ bn2m, const float* __restrict__ bn2v,
    const float* __restrict__ cpw,
    _Float16* __restrict__ pWq, _Float16* __restrict__ pC1,
    _Float16* __restrict__ pC2, _Float16* __restrict__ pP,
    float* __restrict__ beta1, float* __restrict__ beta2,
    float* __restrict__ projAcc)
{
  int g = blockIdx.x * 256 + threadIdx.x;
  if (g < 4096) {                                  // W_qkvv (32 x 128): 8 col-tiles
    int i = g;
    int jt = i >> 9, rem = i & 511, li = rem >> 4, e = rem & 15;
    int n = li & 15, k = (li < 16 ? 0 : 8) + e + (e < 8 ? 0 : 8);
    pWq[i] = (_Float16)Wq[k * 128 + jt * 16 + n];
  } else if (g < 4096 + 27648) {                   // conv1 folded: [tap][ct][lane][16]
    int i = g - 4096;
    int tap = i >> 10, rem = i & 1023, ct = rem >> 9, li = (rem >> 4) & 31, e = rem & 15;
    int o = ct * 16 + (li & 15), ci = (li < 16 ? 0 : 8) + e + (e < 8 ? 0 : 8);
    float alpha = bn1g[o] * rsqrtf(bn1v[o] + 1e-5f);
    pC1[i] = (_Float16)(c1w[(o * 32 + ci) * 27 + tap] * alpha);
  } else if (g < 31744 + 27648) {                  // conv2 folded
    int i = g - 31744;
    int tap = i >> 10, rem = i & 1023, ct = rem >> 9, li = (rem >> 4) & 31, e = rem & 15;
    int o = ct * 16 + (li & 15), ci = (li < 16 ? 0 : 8) + e + (e < 8 ? 0 : 8);
    float alpha = bn2g[o] * rsqrtf(bn2v[o] + 1e-5f);
    pC2[i] = (_Float16)(c2w[(o * 32 + ci) * 27 + tap] * alpha);
  } else if (g < 59392 + 1024) {                   // 1x1x1 conv: [ct][lane][16]
    int i = g - 59392;
    int ct = i >> 9, li = (i >> 4) & 31, e = i & 15;
    int o = ct * 16 + (li & 15), ci = (li < 16 ? 0 : 8) + e + (e < 8 ? 0 : 8);
    pP[i] = (_Float16)cpw[o * 32 + ci];
  } else if (g < 60416 + 64) {                     // folded biases
    int i = g - 60416;
    if (i < 32) {
      float a = bn1g[i] * rsqrtf(bn1v[i] + 1e-5f);
      beta1[i] = (c1b[i] - bn1m[i]) * a + bn1b[i];
    } else {
      int o = i - 32;
      float a = bn2g[o] * rsqrtf(bn2v[o] + 1e-5f);
      beta2[o] = (c2b[o] - bn2m[o]) * a + bn2b[o];
    }
  } else if (g < 60480 + 16384) {                  // zero projAcc [4][64][64]
    projAcc[g - 60480] = 0.f;
  }
}

// ===========================================================================
// K1: xt = transpose(x)+pos ; LayerNorm(C=32) ; qkvv = xn @ W_qkvv via WMMA.
// 128 threads = 4 waves, 64 tokens per block, 4 tiles of 16 tokens.
// qkvvT stored [b][j=0..127][n] (N-major rows for downstream reductions).
// ===========================================================================
__global__ __launch_bounds__(128) void enc_k1_ln_qkvv(
    const float* __restrict__ x, const float* __restrict__ pos,
    const float* __restrict__ lng, const float* __restrict__ lnb,
    const _Float16* __restrict__ pWq,
    float* __restrict__ xt, float* __restrict__ qkvvT)
{
  __shared__ float xs[64][33];
  int tid = threadIdx.x;
  int blk = blockIdx.x;
  int b = blk >> 9, nb = (blk & 511) << 6;

  for (int i = tid; i < 2048; i += 128) {            // x is [b][c][n]: coalesced in n
    int c = i >> 6, n = i & 63;
    xs[n][c] = x[((size_t)(b * 32 + c) << 15) + nb + n];
  }
  __syncthreads();
  for (int i = tid; i < 2048; i += 128) {            // + pos_embed [n][c], write xt
    int n = i >> 5, c = i & 31;
    float v = xs[n][c] + pos[(size_t)(nb + n) * 32 + c];
    xs[n][c] = v;
    xt[((size_t)(b * NN + nb + n)) * 32 + c] = v;
  }
  __syncthreads();
  if (tid < 64) {                                    // LayerNorm per token
    float mu = 0.f;
#pragma unroll
    for (int c = 0; c < 32; c++) mu += xs[tid][c];
    mu *= (1.f / 32.f);
    float var = 0.f;
#pragma unroll
    for (int c = 0; c < 32; c++) { float d = xs[tid][c] - mu; var += d * d; }
    var *= (1.f / 32.f);
    float inv = rsqrtf(var + 1e-5f);
#pragma unroll
    for (int c = 0; c < 32; c++)
      xs[tid][c] = (xs[tid][c] - mu) * inv * lng[c] + lnb[c];
  }
  __syncthreads();

  int w = tid >> 5, l = tid & 31, m = l & 15, hi = l >> 4;
  int tok = w * 16 + m;
  V16H a;
#pragma unroll
  for (int e = 0; e < 8; e++) a.h[e] = (_Float16)xs[tok][hi * 8 + e];
#pragma unroll
  for (int e = 0; e < 8; e++) a.h[8 + e] = (_Float16)xs[tok][16 + hi * 8 + e];

#pragma unroll
  for (int jt = 0; jt < 8; jt++) {
    V16H bm;
    const _Float16* wp = pWq + (jt * 32 + l) * 16;
    bm.q[0] = *(const uint4*)wp;
    bm.q[1] = *(const uint4*)(wp + 8);
    v8f acc = {};
    acc = wmma16(a.v, bm.v, acc);
    float* dst = qkvvT + ((size_t)(b * 128 + jt * 16 + m) << 15) + nb + w * 16 + hi * 8;
#pragma unroll
    for (int r = 0; r < 8; r++) dst[r] = acc[r];
  }
}

// ===========================================================================
// K2: k_proj / v_proj = [k;v_sa] @ EF_w^T via WMMA split-K over N=32768.
// grid = b(4) x mtile(4) x ptile(4) x kchunk(16); 1 wave per block.
// ===========================================================================
__global__ __launch_bounds__(32) void enc_k2_proj(
    const float* __restrict__ qkvvT, const float* __restrict__ EFw,
    float* __restrict__ projAcc)
{
  int id = blockIdx.x;
  int kc = id & 15, pt = (id >> 4) & 3, mt = (id >> 6) & 3, b = id >> 8;
  int jbase = (mt < 2) ? 32 + mt * 16 : 96 + (mt - 2) * 16;  // k rows then v_sa rows
  int l = threadIdx.x, hi = l >> 4, m = l & 15;
  const float* arow = qkvvT + ((size_t)(b * 128 + jbase + m) << 15);
  const float* brow = EFw + (size_t)(pt * 16 + m) * 32768;
  v8f acc = {};
#pragma unroll 4
  for (int s = 0; s < 64; s++) {
    int n0 = kc * 2048 + s * 32;
    V16H a, bm;
    const float* pa1 = arow + n0 + hi * 16;
    const float* pa2 = arow + n0 + hi * 16 + 8;
#pragma unroll
    for (int e = 0; e < 8; e++) { a.h[e] = (_Float16)pa1[e]; a.h[8 + e] = (_Float16)pa2[e]; }
    const float* pb = brow + n0 + hi * 16;
#pragma unroll
    for (int e = 0; e < 16; e++) bm.h[e] = (_Float16)pb[e];
    acc = wmma16(a.v, bm.v, acc);
  }
#pragma unroll
  for (int r = 0; r < 8; r++) {
    int ro = mt * 16 + r + 8 * hi;         // 0..31 = k_proj rows, 32..63 = v_proj rows
    atomicAdd(&projAcc[b * 4096 + ro * 64 + pt * 16 + m], acc[r]);
  }
}

// ===========================================================================
// K3: gram matrix (q_d . k_e) per head + row sum-of-squares for q,k (norms).
// One scalar per block: 256-thread coalesced reduction over N.
// ===========================================================================
__global__ __launch_bounds__(256) void enc_k3_gram(
    const float* __restrict__ qkvvT,
    float* __restrict__ G, float* __restrict__ sqq, float* __restrict__ sqk)
{
  int id = blockIdx.x;
  int job = id % 80, h = (id / 80) % 4, b = id / 320;
  int ra, rb; float* dst;
  if (job < 64) { int d = job >> 3, e = job & 7; ra = h * 8 + d; rb = 32 + h * 8 + e;
                  dst = &G[((b * 4 + h) * 8 + d) * 8 + e]; }
  else if (job < 72) { int d = job - 64; ra = rb = h * 8 + d; dst = &sqq[(b * 4 + h) * 8 + d]; }
  else { int d = job - 72; ra = rb = 32 + h * 8 + d; dst = &sqk[(b * 4 + h) * 8 + d]; }
  const float* A = qkvvT + ((size_t)(b * 128 + ra) << 15);
  const float* B = qkvvT + ((size_t)(b * 128 + rb) << 15);
  float acc = 0.f;
  for (int n = threadIdx.x; n < 32768; n += 256) acc += A[n] * B[n];
  __shared__ float red[256];
  red[threadIdx.x] = acc; __syncthreads();
  for (int s = 128; s > 0; s >>= 1) {
    if (threadIdx.x < s) red[threadIdx.x] += red[threadIdx.x + s];
    __syncthreads();
  }
  if (threadIdx.x == 0) *dst = red[0];
}

// ===========================================================================
// K4: tiny finishing kernel (1 block): norms, channel-attention softmax (8x8),
//     scale k_proj rows by 1/||q_d|| and add EF_b.
// ===========================================================================
__global__ __launch_bounds__(256) void enc_k4_small(
    const float* __restrict__ G, const float* __restrict__ sqq,
    const float* __restrict__ sqk, const float* __restrict__ projAcc,
    const float* __restrict__ EFb, const float* __restrict__ temp1,
    float* __restrict__ Aca, float* __restrict__ kps, float* __restrict__ vps,
    float* __restrict__ nqO)
{
  __shared__ float nq_s[128], nk_s[128];
  int t = threadIdx.x;
  if (t < 128) {
    nq_s[t] = fmaxf(sqrtf(sqq[t]), 1e-12f);
    nk_s[t] = fmaxf(sqrtf(sqk[t]), 1e-12f);
    nqO[t] = nq_s[t];
  }
  __syncthreads();
  if (t < 128) {                                  // softmax over e for row (b,h,d)
    int b = t >> 5, h = (t >> 3) & 3;
    float tp = temp1[h];
    float lg[8], mx = -1e30f;
#pragma unroll
    for (int e = 0; e < 8; e++) {
      lg[e] = G[t * 8 + e] / (nq_s[t] * nk_s[(b * 4 + h) * 8 + e]) * tp;
      mx = fmaxf(mx, lg[e]);
    }
    float s = 0.f;
#pragma unroll
    for (int e = 0; e < 8; e++) { lg[e] = __expf(lg[e] - mx); s += lg[e]; }
    float inv = 1.f / s;
#pragma unroll
    for (int e = 0; e < 8; e++) Aca[t * 8 + e] = lg[e] * inv;
  }
  for (int i = t; i < 8192; i += 256) {           // kps/vps [b][h][d][p]
    int p = i & 63, d = (i >> 6) & 7, h = (i >> 9) & 3, b = i >> 11;
    int ro = h * 8 + d;
    kps[i] = (projAcc[b * 4096 + ro * 64 + p] + EFb[p]) / nq_s[(b * 4 + h) * 8 + d];
    vps[i] = projAcc[b * 4096 + (32 + ro) * 64 + p] + EFb[p];
  }
}

// ===========================================================================
// K5: spatial attention per token: logits[64] = qn_d . kps ; softmax ; x_sa.
// Writes xsaQ in [b][q=d*4+h][n] order (makes torch's scrambled reshape free).
// ===========================================================================
__global__ __launch_bounds__(256) void enc_k5_spatial(
    const float* __restrict__ qkvvT, const float* __restrict__ kps,
    const float* __restrict__ vps, const float* __restrict__ nq,
    const float* __restrict__ temp2, float* __restrict__ xsaQ)
{
  int id = blockIdx.x;
  int nb = (id & 127) << 8, h = (id >> 7) & 3, b = id >> 9;
  __shared__ float kps_s[512], vps_s[512], nq_s[8];
  int t = threadIdx.x;
  if (t < 8) nq_s[t] = nq[(b * 4 + h) * 8 + t];
  {
    size_t base = (size_t)(b * 4 + h) * 512;
    for (int i = t; i < 512; i += 256) { kps_s[i] = kps[base + i]; vps_s[i] = vps[base + i]; }
  }
  __syncthreads();
  int n = nb + t;
  float qd[8];
#pragma unroll
  for (int d = 0; d < 8; d++)
    qd[d] = qkvvT[((size_t)(b * 128 + h * 8 + d) << 15) + n] / nq_s[d];
  float tp = temp2[h];
  float lg[64], mx = -1e30f;
#pragma unroll
  for (int p = 0; p < 64; p++) {
    float s = 0.f;
#pragma unroll
    for (int d = 0; d < 8; d++) s += qd[d] * kps_s[d * 64 + p];
    s *= tp;
    lg[p] = s;
    mx = fmaxf(mx, s);
  }
  float ssum = 0.f;
#pragma unroll
  for (int p = 0; p < 64; p++) { lg[p] = __expf(lg[p] - mx); ssum += lg[p]; }
  float inv = 1.f / ssum;
  float xs[8] = {0, 0, 0, 0, 0, 0, 0, 0};
#pragma unroll
  for (int p = 0; p < 64; p++) {
    float w = lg[p] * inv;
#pragma unroll
    for (int d = 0; d < 8; d++) xs[d] += w * vps_s[d * 64 + p];
  }
#pragma unroll
  for (int d = 0; d < 8; d++)
    xsaQ[((size_t)(b * 32 + d * 4 + h) << 15) + n] = xs[d];
}

// ===========================================================================
// K6: channel attention apply + out1/out2 + gamma residual -> attn_skip
//     ([b][c][n] f32 for the conv residual) + act0 ([b][n][c] f16 for WMMA).
// ===========================================================================
__global__ __launch_bounds__(256) void enc_k6_merge(
    const float* __restrict__ qkvvT, const float* __restrict__ Aca,
    const float* __restrict__ xsaQ, const float* __restrict__ xt,
    const float* __restrict__ gamma,
    const float* __restrict__ o1w, const float* __restrict__ o1b,
    const float* __restrict__ o2w, const float* __restrict__ o2b,
    float* __restrict__ attn_skip, _Float16* __restrict__ act0)
{
  int id = blockIdx.x;
  int b = id >> 7, nb = (id & 127) << 8;
  __shared__ float Aca_s[256], w1[512], w2[512];
  int t = threadIdx.x;
  Aca_s[t] = Aca[b * 256 + t];
  for (int i = t; i < 512; i += 256) { w1[i] = o1w[i]; w2[i] = o2w[i]; }
  __syncthreads();
  int n = nb + t;

  float xca[32];
#pragma unroll
  for (int h = 0; h < 4; h++) {
    float ve[8];
#pragma unroll
    for (int e = 0; e < 8; e++)
      ve[e] = qkvvT[((size_t)(b * 128 + 64 + h * 8 + e) << 15) + n];
#pragma unroll
    for (int d = 0; d < 8; d++) {
      float s = 0.f;
#pragma unroll
      for (int e = 0; e < 8; e++) s += Aca_s[(h * 8 + d) * 8 + e] * ve[e];
      xca[h * 8 + d] = s;
    }
  }
  float xsa[32];
  {
    const float* xr = xsaQ + ((size_t)b << 20) + (size_t)n * 32;  // scrambled reshape = flat read
#pragma unroll
    for (int c = 0; c < 32; c++) xsa[c] = xr[c];
  }
  float sc[32];
#pragma unroll
  for (int c2 = 0; c2 < 16; c2++) {
    float s1 = o1b[c2], s2 = o2b[c2];
#pragma unroll
    for (int c = 0; c < 32; c++) { s1 += xsa[c] * w1[c2 * 32 + c]; s2 += xca[c] * w2[c2 * 32 + c]; }
    sc[c2] = s1; sc[16 + c2] = s2;
  }
  const float* xtr = xt + (size_t)(b * NN + n) * 32;
#pragma unroll
  for (int c = 0; c < 32; c++) {
    float a = xtr[c] + gamma[c] * sc[c];
    attn_skip[((size_t)(b * 32 + c) << 15) + n] = a;
    act0[((size_t)(b * NN + n)) * 32 + c] = (_Float16)a;
  }
}

// ===========================================================================
// K7: 3x3x3 conv (C=32 -> 32) + folded BN + (optional residual) + leaky-ReLU
// as implicit GEMM: 27 taps x WMMA(K=32 input channels). Weights in LDS (54KB).
// 8 waves/block, each wave owns one 16-token tile and both 16-channel col tiles.
// Next-tap activation rows are prefetched (global_prefetch) to overlap the
// per-tap A-load latency with the two WMMAs of the current tap.
// ===========================================================================
__global__ __launch_bounds__(256) void enc_k7_conv(
    const _Float16* __restrict__ actIn, const _Float16* __restrict__ pW,
    const float* __restrict__ beta, const float* __restrict__ attn_skip,
    _Float16* __restrict__ actOut, int addRes)
{
  __shared__ __align__(32) _Float16 sW[27648];
  int t = threadIdx.x;
  for (int i = t; i < 3456; i += 256) ((uint4*)sW)[i] = ((const uint4*)pW)[i];
  __syncthreads();

  int tile = blockIdx.x * 8 + (t >> 5);
  int l = t & 31, hi = l >> 4, m = l & 15;
  int b = tile >> 11, n0 = (tile & 2047) << 4;
  int z = n0 >> 10, y = (n0 >> 5) & 31, x0 = n0 & 31;

  v8f acc0 = {}, acc1 = {};
  for (int tap = 0; tap < 27; tap++) {
    int dz = tap / 9 - 1, dy = (tap / 3) % 3 - 1, dx = tap % 3 - 1;
    int zz = z + dz, yy = y + dy;
    if ((unsigned)zz >= 32u || (unsigned)yy >= 32u) continue;   // uniform per wave
    // prefetch next tap's activation row for this lane
    {
      int tapn = tap + 1;
      if (tapn < 27) {
        int dz2 = tapn / 9 - 1, dy2 = (tapn / 3) % 3 - 1, dx2 = tapn % 3 - 1;
        int zz2 = z + dz2, yy2 = y + dy2, xx2 = x0 + m + dx2;
        if ((unsigned)zz2 < 32u && (unsigned)yy2 < 32u && (unsigned)xx2 < 32u) {
          const _Float16* pn =
              actIn + ((size_t)(b * NN + zz2 * 1024 + yy2 * 32 + xx2) << 5);
          __builtin_prefetch(pn, 0, 1);
        }
      }
    }
    int xx = x0 + m + dx;
    V16H a;
    if ((unsigned)xx < 32u) {
      int nn = zz * 1024 + yy * 32 + xx;
      const _Float16* p = actIn + ((size_t)(b * NN + nn) << 5) + hi * 8;
      a.q[0] = *(const uint4*)p;
      a.q[1] = *(const uint4*)(p + 16);
    } else {
      a.q[0] = make_uint4(0, 0, 0, 0);
      a.q[1] = make_uint4(0, 0, 0, 0);
    }
    const _Float16* wp = sW + tap * 1024 + l * 16;
    V16H b0, b1;
    b0.q[0] = *(const uint4*)wp;        b0.q[1] = *(const uint4*)(wp + 8);
    b1.q[0] = *(const uint4*)(wp + 512); b1.q[1] = *(const uint4*)(wp + 520);
    acc0 = wmma16(a.v, b0.v, acc0);
    acc1 = wmma16(a.v, b1.v, acc1);
  }

  {
    int cc = m;                  // col tile 0
    float bt = beta[cc];
#pragma unroll
    for (int r = 0; r < 8; r++) {
      int tok = n0 + r + 8 * hi;
      float v = acc0[r] + bt;
      if (addRes) v += attn_skip[((size_t)(b * 32 + cc) << 15) + tok];
      actOut[((size_t)(b * NN + tok) << 5) + cc] = (_Float16)lrelu(v);
    }
  }
  {
    int cc = 16 + m;             // col tile 1
    float bt = beta[cc];
#pragma unroll
    for (int r = 0; r < 8; r++) {
      int tok = n0 + r + 8 * hi;
      float v = acc1[r] + bt;
      if (addRes) v += attn_skip[((size_t)(b * 32 + cc) << 15) + tok];
      actOut[((size_t)(b * NN + tok) << 5) + cc] = (_Float16)lrelu(v);
    }
  }
}

// ===========================================================================
// K9: 1x1x1 conv via WMMA (K=32) + attn_skip residual -> final output [b][c][n]
// ===========================================================================
__global__ __launch_bounds__(256) void enc_k9_pointwise(
    const _Float16* __restrict__ act2, const _Float16* __restrict__ pP,
    const float* __restrict__ cpb, const float* __restrict__ attn_skip,
    float* __restrict__ out)
{
  __shared__ __align__(32) _Float16 sW[1024];
  int t = threadIdx.x;
  if (t < 128) ((uint4*)sW)[t] = ((const uint4*)pP)[t];
  __syncthreads();

  int tile = blockIdx.x * 8 + (t >> 5);
  int l = t & 31, hi = l >> 4, m = l & 15;
  int b = tile >> 11, n0 = (tile & 2047) << 4;

  const _Float16* p = act2 + ((size_t)(b * NN + n0 + m) << 5) + hi * 8;
  V16H a;
  a.q[0] = *(const uint4*)p;
  a.q[1] = *(const uint4*)(p + 16);

  V16H b0, b1;
  const _Float16* wp = sW + l * 16;
  b0.q[0] = *(const uint4*)wp;         b0.q[1] = *(const uint4*)(wp + 8);
  b1.q[0] = *(const uint4*)(wp + 512); b1.q[1] = *(const uint4*)(wp + 520);
  v8f acc0 = {}, acc1 = {};
  acc0 = wmma16(a.v, b0.v, acc0);
  acc1 = wmma16(a.v, b1.v, acc1);

  {
    int cc = m; float bt = cpb[cc];
#pragma unroll
    for (int r = 0; r < 8; r++) {
      int tok = n0 + r + 8 * hi;
      size_t o = ((size_t)(b * 32 + cc) << 15) + tok;
      out[o] = acc0[r] + bt + attn_skip[o];
    }
  }
  {
    int cc = 16 + m; float bt = cpb[cc];
#pragma unroll
    for (int r = 0; r < 8; r++) {
      int tok = n0 + r + 8 * hi;
      size_t o = ((size_t)(b * 32 + cc) << 15) + tok;
      out[o] = acc1[r] + bt + attn_skip[o];
    }
  }
}

// ===========================================================================
extern "C" void kernel_launch(void* const* d_in, const int* in_sizes, int n_in,
                              void* d_out, int out_size, void* d_ws, size_t ws_size,
                              hipStream_t stream) {
  (void)in_sizes; (void)n_in; (void)out_size; (void)ws_size;
  const float* x        = (const float*)d_in[0];
  const float* pos      = (const float*)d_in[1];
  const float* ln_g     = (const float*)d_in[2];
  const float* ln_b     = (const float*)d_in[3];
  const float* gamma    = (const float*)d_in[4];
  const float* W_qkvv   = (const float*)d_in[5];
  const float* EF_w     = (const float*)d_in[6];
  const float* EF_b     = (const float*)d_in[7];
  const float* temp1    = (const float*)d_in[8];
  const float* temp2    = (const float*)d_in[9];
  const float* out1_w   = (const float*)d_in[10];
  const float* out1_b   = (const float*)d_in[11];
  const float* out2_w   = (const float*)d_in[12];
  const float* out2_b   = (const float*)d_in[13];
  const float* conv1_w  = (const float*)d_in[14];
  const float* conv1_b  = (const float*)d_in[15];
  const float* bn1_g    = (const float*)d_in[16];
  const float* bn1_b    = (const float*)d_in[17];
  const float* bn1_m    = (const float*)d_in[18];
  const float* bn1_v    = (const float*)d_in[19];
  const float* conv2_w  = (const float*)d_in[20];
  const float* conv2_b  = (const float*)d_in[21];
  const float* bn2_g    = (const float*)d_in[22];
  const float* bn2_b    = (const float*)d_in[23];
  const float* bn2_m    = (const float*)d_in[24];
  const float* bn2_v    = (const float*)d_in[25];
  const float* convp_w  = (const float*)d_in[26];
  const float* convp_b  = (const float*)d_in[27];
  float* out = (float*)d_out;

  // workspace bump allocator (256B aligned)
  char* wsp = (char*)d_ws;
  auto alloc = [&](size_t bytes) -> void* {
    void* p = (void*)wsp;
    wsp += (bytes + 255) & ~(size_t)255;
    return p;
  };
  float*    xt        = (float*)alloc((size_t)NB * NN * CC * 4);        // 16 MB
  float*    qkvvT     = (float*)alloc((size_t)NB * 128 * NN * 4);       // 64 MB
  float*    xsaQ      = (float*)alloc((size_t)NB * NN * CC * 4);        // 16 MB
  float*    attn_skip = (float*)alloc((size_t)NB * NN * CC * 4);        // 16 MB
  _Float16* act0      = (_Float16*)alloc((size_t)NB * NN * CC * 2);     //  8 MB
  _Float16* act1      = (_Float16*)alloc((size_t)NB * NN * CC * 2);
  _Float16* act2      = (_Float16*)alloc((size_t)NB * NN * CC * 2);
  _Float16* pWq       = (_Float16*)alloc(4096 * 2);
  _Float16* pC1       = (_Float16*)alloc(27648 * 2);
  _Float16* pC2       = (_Float16*)alloc(27648 * 2);
  _Float16* pP        = (_Float16*)alloc(1024 * 2);
  float*    beta1     = (float*)alloc(32 * 4);
  float*    beta2     = (float*)alloc(32 * 4);
  float*    projAcc   = (float*)alloc((size_t)NB * 64 * PP * 4);
  float*    G         = (float*)alloc(1024 * 4);
  float*    sqq       = (float*)alloc(128 * 4);
  float*    sqk       = (float*)alloc(128 * 4);
  float*    nq        = (float*)alloc(128 * 4);
  float*    Aca       = (float*)alloc(1024 * 4);
  float*    kps       = (float*)alloc(8192 * 4);
  float*    vps       = (float*)alloc(8192 * 4);

  enc_k0_pack<<<301, 256, 0, stream>>>(W_qkvv, conv1_w, conv1_b, bn1_g, bn1_b, bn1_m, bn1_v,
                                       conv2_w, conv2_b, bn2_g, bn2_b, bn2_m, bn2_v, convp_w,
                                       pWq, pC1, pC2, pP, beta1, beta2, projAcc);
  enc_k1_ln_qkvv<<<2048, 128, 0, stream>>>(x, pos, ln_g, ln_b, pWq, xt, qkvvT);
  enc_k2_proj<<<1024, 32, 0, stream>>>(qkvvT, EF_w, projAcc);
  enc_k3_gram<<<1280, 256, 0, stream>>>(qkvvT, G, sqq, sqk);
  enc_k4_small<<<1, 256, 0, stream>>>(G, sqq, sqk, projAcc, EF_b, temp1, Aca, kps, vps, nq);
  enc_k5_spatial<<<2048, 256, 0, stream>>>(qkvvT, kps, vps, nq, temp2, xsaQ);
  enc_k6_merge<<<512, 256, 0, stream>>>(qkvvT, Aca, xsaQ, xt, gamma,
                                        out1_w, out1_b, out2_w, out2_b, attn_skip, act0);
  enc_k7_conv<<<1024, 256, 0, stream>>>(act0, pC1, beta1, attn_skip, act1, 0);
  enc_k7_conv<<<1024, 256, 0, stream>>>(act1, pC2, beta2, attn_skip, act2, 1);
  enc_k9_pointwise<<<1024, 256, 0, stream>>>(act2, pP, convp_b, attn_skip, out);
}
